// EnhancedProbeReadoutModel_90941637526150
// MI455X (gfx1250) — compile-verified, matrix operand order checked by reading, not run on hardware
//
#include <hip/hip_runtime.h>
#include <hip/hip_bf16.h>
#include <math.h>

typedef __attribute__((ext_vector_type(16))) _Float16 v16h;
typedef __attribute__((ext_vector_type(8)))  _Float16 v8h;
typedef __attribute__((ext_vector_type(8)))  float    v8f;

#define N_ATOMS   20000
#define N_PROBES  100000
#define N_EDGES   300000
#define D_S       256
#define D_VCH     128
#define D_RBF     20
#define DIM_IN    532
#define KPAD1     544
#define KC1       17          // 544/32 k-chunks for layer 1
#define MSG_H     256
#define KC2       8           // 256/32 k-chunks for layer 2 / readout
#define OUT_H     128
#define CUTOFFV   4.0f
#define INV_SQRT20 0.22360679774997896f
#define TILE_E    32          // edges per block (2 WMMA M-tiles)
#define TILE_P    32          // probes per block (2 WMMA M-tiles)

// ---- workspace layout (all offsets 256B aligned) ----
#define OFF_MP    0ul
#define SZ_MP     ((size_t)N_PROBES * MSG_H * 4)          // 102,400,000
#define OFF_W1P   (OFF_MP + SZ_MP)                        // 544*256 f16
#define OFF_W2P   (OFF_W1P + (size_t)KPAD1 * MSG_H * 2)   // 256*256 f16
#define OFF_WO1P  (OFF_W2P + (size_t)MSG_H * MSG_H * 2)   // 256*128 f16
#define OFF_RBF   (OFF_WO1P + (size_t)MSG_H * OUT_H * 2)  // mean[20], invstd[20]

__device__ __forceinline__ float silu_f(float x) {
    return x * (1.0f / (1.0f + __expf(-x)));
}

// ---------------- prep kernels ----------------

__global__ void zero4_kernel(float4* __restrict__ p, long n4) {
    long i = (long)blockIdx.x * blockDim.x + threadIdx.x;
    long stride = (long)gridDim.x * blockDim.x;
    float4 z = make_float4(0.f, 0.f, 0.f, 0.f);
    for (; i < n4; i += stride) p[i] = z;
}

// RBF normalization stats: mean and 1/std (ddof=1) over 4000 linspace samples.
__global__ void rbf_stats_kernel(float* __restrict__ stats) {
    int t = threadIdx.x;
    if (t >= D_RBF) return;
    const float step = CUTOFFV / 19.0f;
    float center = t * step;
    float s = 0.f, s2 = 0.f;
    for (int k = 1; k <= 4000; ++k) {
        float r = (float)k * (CUTOFFV / 4000.0f);
        float u = (r - center) / step;
        float b = __expf(-u * u) * (1.0f / 1.12f);
        s += b; s2 += b * b;
    }
    float mean = s * (1.0f / 4000.0f);
    float var  = (s2 - 4000.0f * mean * mean) * (1.0f / 3999.0f);
    stats[t] = mean;
    stats[D_RBF + t] = rsqrtf(var);
}

// Pack a row-major KxN f32 weight into WMMA B-fragment order (f16), K padded
// to KC*32 with zeros. Output element (((nt*KC + kc)*32 + lane)*16 + j) holds
// W[k][n] with kb=(lane>>4)*8, k = kc*32 + (j<8 ? kb+j : 16+kb+j-8),
// n = nt*16 + (lane&15). Each lane then loads its 16 halves (32 bytes) contiguously.
__global__ void pack_b_kernel(const float* __restrict__ W, _Float16* __restrict__ out,
                              int K, int KC, int N) {
    int tid = blockIdx.x * blockDim.x + threadIdx.x;
    int total = (N / 16) * KC * 512;
    if (tid >= total) return;
    int j    = tid & 15;
    int lane = (tid >> 4) & 31;
    int kc   = (tid >> 9) % KC;
    int nt   = tid / (KC * 512);
    int kb   = (lane >> 4) * 8;
    int k    = kc * 32 + ((j < 8) ? (kb + j) : (16 + kb + (j - 8)));
    int n    = nt * 16 + (lane & 15);
    float v  = (k < K) ? W[(long)k * N + n] : 0.0f;
    out[tid] = (_Float16)v;
}

// ---------------- fused edge kernel ----------------
// One block = 32 edges (2 WMMA M-tiles). Phase A builds the 532-wide message
// in LDS (f32), LayerNorms it, casts to f16 (K padded to 544). Phase B: two
// WMMA GEMMs (each wave: 2 M-tiles x 2 N-tiles = 4 accumulators) with SiLU in
// between, then envelope-scaled atomic scatter into m_p.

__launch_bounds__(256)
__global__ void edge_kernel(
    const float* __restrict__ atom_xyz, const float* __restrict__ probe_xyz,
    const float* __restrict__ cell,     const float* __restrict__ edisp,
    const float* __restrict__ S_final,  const float* __restrict__ V_final,
    const float* __restrict__ ln_gamma, const float* __restrict__ ln_beta,
    const _Float16* __restrict__ W1p,   const float* __restrict__ b1,
    const _Float16* __restrict__ W2p,   const float* __restrict__ b2,
    const float* __restrict__ rbf_stats,
    const int* __restrict__ edge_src,   const int* __restrict__ edge_dst,
    float* __restrict__ mp)
{
    __shared__ __attribute__((aligned(16))) float     msg32[TILE_E * DIM_IN];
    __shared__ __attribute__((aligned(16))) _Float16  msgh [TILE_E * KPAD1];
    __shared__ __attribute__((aligned(16))) _Float16  hbuf [TILE_E * MSG_H];
    __shared__ float rhat_s[TILE_E][3];
    __shared__ float env_s[TILE_E];
    __shared__ int   dst_s[TILE_E];
    __shared__ int   src_s[TILE_E];

    const int tid  = threadIdx.x;
    const int lane = tid & 31;
    const int wave = tid >> 5;
    const int e0   = blockIdx.x * TILE_E;

    // ---- Phase A1: geometry, envelope, RBF (wave 0: one lane per edge) ----
    if (tid < TILE_E) {
        int e = e0 + tid;
        int s = edge_src[e];
        int d_idx = edge_dst[e];
        src_s[tid] = s;
        dst_s[tid] = d_idx;
        float edx = edisp[e*3+0], edy = edisp[e*3+1], edz = edisp[e*3+2];
        float dpx = edx*cell[0] + edy*cell[3] + edz*cell[6];
        float dpy = edx*cell[1] + edy*cell[4] + edz*cell[7];
        float dpz = edx*cell[2] + edy*cell[5] + edz*cell[8];
        float vx = probe_xyz[(long)d_idx*3+0] - (atom_xyz[(long)s*3+0] + dpx);
        float vy = probe_xyz[(long)d_idx*3+1] - (atom_xyz[(long)s*3+1] + dpy);
        float vz = probe_xyz[(long)d_idx*3+2] - (atom_xyz[(long)s*3+2] + dpz);
        float d = fmaxf(sqrtf(vx*vx + vy*vy + vz*vz), 1e-8f);
        float inv = 1.0f / d;
        rhat_s[tid][0] = vx * inv; rhat_s[tid][1] = vy * inv; rhat_s[tid][2] = vz * inv;
        float x  = d * (1.0f / CUTOFFV);
        float x2 = x * x;
        float x5 = x2 * x2 * x;
        float env = 1.0f - 21.0f*x5 + 35.0f*x5*x - 15.0f*x5*x2;
        env_s[tid] = (d < CUTOFFV) ? env : 0.0f;
        const float step = CUTOFFV / 19.0f;
        #pragma unroll
        for (int i = 0; i < D_RBF; ++i) {
            float u = (d - (float)i * step) / step;
            float ev = __expf(-u * u) * (1.0f / 1.12f);
            msg32[tid * DIM_IN + D_S + D_VCH + i] =
                (ev - rbf_stats[i]) * rbf_stats[D_RBF + i];
        }
    }
    __syncthreads();

    // ---- Phase A2: gather S_i (32 rows x 256) ----
    for (int idx = tid; idx < TILE_E * D_S; idx += 256) {
        int r = idx >> 8, c = idx & 255;
        msg32[r * DIM_IN + c] = S_final[(long)src_s[r] * D_S + c];
    }
    // ---- Phase A3: gather V_i, compute |v| and r_hat . v ----
    for (int idx = tid; idx < TILE_E * D_VCH; idx += 256) {
        int r = idx >> 7, ch = idx & 127;
        const float* vp = &V_final[(long)src_s[r] * (D_VCH * 3) + ch * 3];
        float x = vp[0], y = vp[1], z = vp[2];
        msg32[r * DIM_IN + D_S + ch] = sqrtf(x*x + y*y + z*z + 1e-8f);
        msg32[r * DIM_IN + D_S + D_VCH + D_RBF + ch] =
            rhat_s[r][0]*x + rhat_s[r][1]*y + rhat_s[r][2]*z;
    }
    __syncthreads();

    // ---- Phase A4: LayerNorm (each wave owns 4 rows) + cast to f16 ----
    #pragma unroll
    for (int rr = 0; rr < 4; ++rr) {
        int r = wave * 4 + rr;
        const float* row = &msg32[r * DIM_IN];
        float s = 0.f, s2 = 0.f;
        for (int k = lane; k < DIM_IN; k += 32) { float v = row[k]; s += v; s2 += v * v; }
        #pragma unroll
        for (int off = 16; off > 0; off >>= 1) {
            s  += __shfl_xor(s,  off, 32);
            s2 += __shfl_xor(s2, off, 32);
        }
        float mean = s * (1.0f / DIM_IN);
        float var  = s2 * (1.0f / DIM_IN) - mean * mean;
        float rstd = rsqrtf(var + 1e-5f);
        for (int k = lane; k < KPAD1; k += 32) {
            float v = (k < DIM_IN) ? ((row[k] - mean) * rstd * ln_gamma[k] + ln_beta[k])
                                   : 0.0f;
            msgh[r * KPAD1 + k] = (_Float16)v;
        }
    }
    __syncthreads();

    const int arow  = lane & 15;          // A-fragment row within M-tile
    const int kb    = (lane >> 4) * 8;    // per-lane K sub-offset
    const int crow0 = (lane >> 4) * 8;    // C-fragment first row within M-tile
    const int ccol  = lane & 15;          // C-fragment column
    const int nt0 = wave * 2, nt1 = wave * 2 + 1;
    const int col0 = nt0 * 16 + ccol, col1 = nt1 * 16 + ccol;

    // ---- GEMM1: h = silu(msg @ W1 + b1), K=544, 2x2 tiles per wave ----
    v8f c00 = {}, c01 = {}, c10 = {}, c11 = {};
    for (int kc = 0; kc < KC1; ++kc) {
        int ab0 = arow * KPAD1 + kc * 32 + kb;
        int ab1 = (16 + arow) * KPAD1 + kc * 32 + kb;
        v8h lo0 = *(const v8h*)&msgh[ab0];
        v8h hi0 = *(const v8h*)&msgh[ab0 + 16];
        v8h lo1 = *(const v8h*)&msgh[ab1];
        v8h hi1 = *(const v8h*)&msgh[ab1 + 16];
        v16h a0 = __builtin_shufflevector(lo0, hi0, 0,1,2,3,4,5,6,7,8,9,10,11,12,13,14,15);
        v16h a1 = __builtin_shufflevector(lo1, hi1, 0,1,2,3,4,5,6,7,8,9,10,11,12,13,14,15);
        v16h bf0 = *(const v16h*)&W1p[((nt0 * KC1 + kc) * 32 + lane) * 16];
        v16h bf1 = *(const v16h*)&W1p[((nt1 * KC1 + kc) * 32 + lane) * 16];
        c00 = __builtin_amdgcn_wmma_f32_16x16x32_f16(false, a0, false, bf0, (short)0, c00, false, false);
        c01 = __builtin_amdgcn_wmma_f32_16x16x32_f16(false, a0, false, bf1, (short)0, c01, false, false);
        c10 = __builtin_amdgcn_wmma_f32_16x16x32_f16(false, a1, false, bf0, (short)0, c10, false, false);
        c11 = __builtin_amdgcn_wmma_f32_16x16x32_f16(false, a1, false, bf1, (short)0, c11, false, false);
    }
    {
        float bb0 = b1[col0], bb1 = b1[col1];
        #pragma unroll
        for (int r = 0; r < 8; ++r) {
            int row0 = crow0 + r, row1 = 16 + crow0 + r;
            hbuf[row0 * MSG_H + col0] = (_Float16)silu_f(c00[r] + bb0);
            hbuf[row0 * MSG_H + col1] = (_Float16)silu_f(c01[r] + bb1);
            hbuf[row1 * MSG_H + col0] = (_Float16)silu_f(c10[r] + bb0);
            hbuf[row1 * MSG_H + col1] = (_Float16)silu_f(c11[r] + bb1);
        }
    }
    __syncthreads();

    // ---- GEMM2: m = silu(h @ W2 + b2) * env; atomic scatter to m_p ----
    v8f d00 = {}, d01 = {}, d10 = {}, d11 = {};
    for (int kc = 0; kc < KC2; ++kc) {
        int ab0 = arow * MSG_H + kc * 32 + kb;
        int ab1 = (16 + arow) * MSG_H + kc * 32 + kb;
        v8h lo0 = *(const v8h*)&hbuf[ab0];
        v8h hi0 = *(const v8h*)&hbuf[ab0 + 16];
        v8h lo1 = *(const v8h*)&hbuf[ab1];
        v8h hi1 = *(const v8h*)&hbuf[ab1 + 16];
        v16h a0 = __builtin_shufflevector(lo0, hi0, 0,1,2,3,4,5,6,7,8,9,10,11,12,13,14,15);
        v16h a1 = __builtin_shufflevector(lo1, hi1, 0,1,2,3,4,5,6,7,8,9,10,11,12,13,14,15);
        v16h bf0 = *(const v16h*)&W2p[((nt0 * KC2 + kc) * 32 + lane) * 16];
        v16h bf1 = *(const v16h*)&W2p[((nt1 * KC2 + kc) * 32 + lane) * 16];
        d00 = __builtin_amdgcn_wmma_f32_16x16x32_f16(false, a0, false, bf0, (short)0, d00, false, false);
        d01 = __builtin_amdgcn_wmma_f32_16x16x32_f16(false, a0, false, bf1, (short)0, d01, false, false);
        d10 = __builtin_amdgcn_wmma_f32_16x16x32_f16(false, a1, false, bf0, (short)0, d10, false, false);
        d11 = __builtin_amdgcn_wmma_f32_16x16x32_f16(false, a1, false, bf1, (short)0, d11, false, false);
    }
    {
        float bb0 = b2[col0], bb1 = b2[col1];
        #pragma unroll
        for (int r = 0; r < 8; ++r) {
            int row0 = crow0 + r, row1 = 16 + crow0 + r;
            float sc0 = env_s[row0] * INV_SQRT20;
            float sc1 = env_s[row1] * INV_SQRT20;
            long base0 = (long)dst_s[row0] * MSG_H;
            long base1 = (long)dst_s[row1] * MSG_H;
            atomicAdd(&mp[base0 + col0], silu_f(d00[r] + bb0) * sc0);
            atomicAdd(&mp[base0 + col1], silu_f(d01[r] + bb1) * sc0);
            atomicAdd(&mp[base1 + col0], silu_f(d10[r] + bb0) * sc1);
            atomicAdd(&mp[base1 + col1], silu_f(d11[r] + bb1) * sc1);
        }
    }
}

// ---------------- probe readout kernel ----------------
// One block = 32 probes: rho = silu(m_p @ Wo1 + bo1) @ Wo2 + bo2

__launch_bounds__(256)
__global__ void probe_kernel(const float* __restrict__ mp,
                             const _Float16* __restrict__ Wo1p,
                             const float* __restrict__ bo1,
                             const float* __restrict__ Wo2,
                             const float* __restrict__ bo2,
                             float* __restrict__ out)
{
    __shared__ __attribute__((aligned(16))) _Float16 ah[TILE_P * MSG_H];
    __shared__ float hb[TILE_P * OUT_H];
    const int tid = threadIdx.x;
    const int lane = tid & 31;
    const int wave = tid >> 5;
    const int p0 = blockIdx.x * TILE_P;

    for (int idx = tid; idx < TILE_P * MSG_H; idx += 256) {
        int r = idx >> 8, c = idx & 255;
        ah[idx] = (_Float16)mp[(long)(p0 + r) * MSG_H + c];
    }
    __syncthreads();

    const int arow  = lane & 15;
    const int kb    = (lane >> 4) * 8;
    const int crow0 = (lane >> 4) * 8;
    const int ccol  = lane & 15;
    const int nt = wave;                  // 8 waves x 1 N-tile = 128 cols
    const int col = nt * 16 + ccol;

    v8f acc0 = {}, acc1 = {};
    for (int kc = 0; kc < KC2; ++kc) {
        int ab0 = arow * MSG_H + kc * 32 + kb;
        int ab1 = (16 + arow) * MSG_H + kc * 32 + kb;
        v8h lo0 = *(const v8h*)&ah[ab0];
        v8h hi0 = *(const v8h*)&ah[ab0 + 16];
        v8h lo1 = *(const v8h*)&ah[ab1];
        v8h hi1 = *(const v8h*)&ah[ab1 + 16];
        v16h a0 = __builtin_shufflevector(lo0, hi0, 0,1,2,3,4,5,6,7,8,9,10,11,12,13,14,15);
        v16h a1 = __builtin_shufflevector(lo1, hi1, 0,1,2,3,4,5,6,7,8,9,10,11,12,13,14,15);
        v16h b = *(const v16h*)&Wo1p[((nt * KC2 + kc) * 32 + lane) * 16];
        acc0 = __builtin_amdgcn_wmma_f32_16x16x32_f16(false, a0, false, b, (short)0, acc0, false, false);
        acc1 = __builtin_amdgcn_wmma_f32_16x16x32_f16(false, a1, false, b, (short)0, acc1, false, false);
    }
    {
        float bb = bo1[col];
        #pragma unroll
        for (int r = 0; r < 8; ++r) {
            hb[(crow0 + r) * OUT_H + col] = silu_f(acc0[r] + bb);
            hb[(16 + crow0 + r) * OUT_H + col] = silu_f(acc1[r] + bb);
        }
    }
    __syncthreads();

    if (tid < TILE_P) {
        float r = bo2[0];
        for (int k = 0; k < OUT_H; ++k) r += hb[tid * OUT_H + k] * Wo2[k];
        out[p0 + tid] = r;
    }
}

// ---------------- launcher ----------------

extern "C" void kernel_launch(void* const* d_in, const int* in_sizes, int n_in,
                              void* d_out, int out_size, void* d_ws, size_t ws_size,
                              hipStream_t stream) {
    const float* atom_xyz  = (const float*)d_in[0];
    const float* probe_xyz = (const float*)d_in[1];
    const float* cell      = (const float*)d_in[2];
    const float* edisp     = (const float*)d_in[3];
    const float* S_final   = (const float*)d_in[4];
    const float* V_final   = (const float*)d_in[5];
    const float* ln_gamma  = (const float*)d_in[6];
    const float* ln_beta   = (const float*)d_in[7];
    const float* W1        = (const float*)d_in[8];
    const float* b1        = (const float*)d_in[9];
    const float* W2        = (const float*)d_in[10];
    const float* b2        = (const float*)d_in[11];
    const float* Wo1       = (const float*)d_in[12];
    const float* bo1       = (const float*)d_in[13];
    const float* Wo2       = (const float*)d_in[14];
    const float* bo2       = (const float*)d_in[15];
    const int*   edge_src  = (const int*)d_in[16];
    const int*   edge_dst  = (const int*)d_in[17];

    char* ws = (char*)d_ws;
    float*     mp    = (float*)(ws + OFF_MP);
    _Float16*  W1p   = (_Float16*)(ws + OFF_W1P);
    _Float16*  W2p   = (_Float16*)(ws + OFF_W2P);
    _Float16*  Wo1p  = (_Float16*)(ws + OFF_WO1P);
    float*     rbfs  = (float*)(ws + OFF_RBF);

    // prep: zero accumulator, RBF stats, fragment-pack weights to f16
    long n4 = (long)N_PROBES * MSG_H / 4;
    zero4_kernel<<<2048, 256, 0, stream>>>((float4*)mp, n4);
    rbf_stats_kernel<<<1, 32, 0, stream>>>(rbfs);
    {
        int tot1 = (MSG_H / 16) * KC1 * 512;   // 139264
        pack_b_kernel<<<(tot1 + 255) / 256, 256, 0, stream>>>(W1, W1p, DIM_IN, KC1, MSG_H);
        int tot2 = (MSG_H / 16) * KC2 * 512;   // 65536
        pack_b_kernel<<<(tot2 + 255) / 256, 256, 0, stream>>>(W2, W2p, MSG_H, KC2, MSG_H);
        int tot3 = (OUT_H / 16) * KC2 * 512;   // 32768
        pack_b_kernel<<<(tot3 + 255) / 256, 256, 0, stream>>>(Wo1, Wo1p, MSG_H, KC2, OUT_H);
    }

    // fused edge pass: 300000 / 32 = 9375 tiles
    edge_kernel<<<N_EDGES / TILE_E, 256, 0, stream>>>(
        atom_xyz, probe_xyz, cell, edisp, S_final, V_final,
        ln_gamma, ln_beta, W1p, b1, W2p, b2, rbfs,
        edge_src, edge_dst, mp);

    // probe readout: 100000 / 32 = 3125 tiles
    probe_kernel<<<N_PROBES / TILE_P, 256, 0, stream>>>(
        mp, Wo1p, bo1, Wo2, bo2, (float*)d_out);
}
